// HybridSlotRung_62801011802691
// MI455X (gfx1250) — matrix operand order repaired and checked by reading, not compiled, not run on hardware
//
#include <hip/hip_runtime.h>
#include <hip/hip_bf16.h>
#include <math.h>

// ---------------------------------------------------------------------------
// HybridSlotRung for MI455X (gfx1250)
//   - GEMMs: v_wmma_f32_16x16x32_bf16 (f32 accumulate)
//   - B panels staged to LDS via global_load_async_to_lds_b128 (ASYNCcnt),
//     double-buffered so staging overlaps the WMMA K-loop.
// ---------------------------------------------------------------------------

#define BB 16384          // batch
#define SS 6              // slots
#define GDIM 1669         // gate feature dim
#define GPAD 1696         // padded to 53*32
#define GLDS 1704         // LDS row stride (halves) for bank spread

typedef __bf16 bf16_t;
typedef __attribute__((ext_vector_type(16))) __bf16 v16bf;
typedef __attribute__((ext_vector_type(8)))  float  v8f;

__device__ __forceinline__ unsigned short f2bf(float f) {
    unsigned int u = __float_as_uint(f);
    unsigned int r = (u + 0x7FFFu + ((u >> 16) & 1u)) >> 16;
    return (unsigned short)r;
}

__device__ __forceinline__ float gelu_exact(float x) {
    return 0.5f * x * (1.0f + erff(x * 0.70710678118654752440f));
}
__device__ __forceinline__ float sigmoidf(float x) {
    return 1.0f / (1.0f + expf(-x));
}

union V16U { v16bf v; uint4 u[2]; };

// A-matrix tile loader (16x32 bf16): two 16B chunks at +0 and +16B.
// Caller passes p = base + row*lda + k0 + ((lane>>4)<<4).
__device__ __forceinline__ v16bf ld_a(const unsigned short* p) {
    V16U t;
    t.u[0] = *reinterpret_cast<const uint4*>(p);
    t.u[1] = *reinterpret_cast<const uint4*>(p + 8);
    return t.v;
}
// B-matrix tile loader (32x16 bf16, N-major / K-contiguous): 16 contiguous bf16.
// Caller passes p = base + n*ldk + k0 + ((lane>>4)<<4).
__device__ __forceinline__ v16bf ld_b(const unsigned short* p) {
    V16U t;
    t.u[0] = *reinterpret_cast<const uint4*>(p);
    t.u[1] = *reinterpret_cast<const uint4*>(p + 8);
    return t.v;
}

__device__ __forceinline__ v8f vzero8() {
    v8f z;
#pragma unroll
    for (int i = 0; i < 8; ++i) z[i] = 0.0f;
    return z;
}

// Async 16B global -> LDS copy (CDNA5 TDM-adjacent async path, ASYNCcnt).
__device__ __forceinline__ void async_g2l_b128(unsigned lds_off, const unsigned short* g) {
    unsigned long long ga = (unsigned long long)(uintptr_t)g;
    asm volatile("global_load_async_to_lds_b128 %0, %1, off"
                 :: "v"(lds_off), "v"(ga) : "memory");
}
__device__ __forceinline__ void wait_asynccnt0() {
    asm volatile("s_wait_asynccnt 0x0" ::: "memory");
}
__device__ __forceinline__ unsigned lds_addr(const void* p) {
    return (unsigned)(uintptr_t)p;   // low 32 bits of LDS-aperture generic addr
}

#define WMMA_BF16(A, Bm, Cm) \
    __builtin_amdgcn_wmma_f32_16x16x32_bf16(false, (A), false, (Bm), (short)0, (Cm), false, false)

// ---------------------------------------------------------------------------
// Packing / conversion kernels
// ---------------------------------------------------------------------------

__global__ __launch_bounds__(256) void k_cvt_bf16(const float* __restrict__ src,
                                                  unsigned short* __restrict__ dst, long n) {
    long i = (long)blockIdx.x * blockDim.x + threadIdx.x;
    if (i < n) dst[i] = f2bf(src[i]);
}

// W1t[n][k] : n<512 -> Wk1[k][n] ; n>=512 -> Wv1[k][n-512]   (ldk = 1024)
__global__ __launch_bounds__(256) void k_pack_w1(const float* __restrict__ Wk1,
                                                 const float* __restrict__ Wv1,
                                                 unsigned short* __restrict__ out) {
    long i = (long)blockIdx.x * blockDim.x + threadIdx.x;
    if (i >= (long)1024 * 1024) return;
    int n = (int)(i >> 10), k = (int)(i & 1023);
    float v = (n < 512) ? Wk1[(long)k * 512 + n] : Wv1[(long)k * 512 + (n - 512)];
    out[i] = f2bf(v);
}

// W2t[n][k] (n<384, ldk=1024): block diagonal [Wk2 | 0 ; 0 | Wv2]
__global__ __launch_bounds__(256) void k_pack_w2(const float* __restrict__ Wk2,
                                                 const float* __restrict__ Wv2,
                                                 unsigned short* __restrict__ out) {
    long i = (long)blockIdx.x * blockDim.x + threadIdx.x;
    if (i >= (long)384 * 1024) return;
    int n = (int)(i >> 10), k = (int)(i & 1023);
    float v = 0.0f;
    if (n < 128) { if (k < 512) v = Wk2[(long)k * 128 + n]; }
    else         { if (k >= 512) v = Wv2[(long)(k - 512) * 256 + (n - 128)]; }
    out[i] = f2bf(v);
}

// Wg1t[(g*384+n)][k] (ldk = GPAD), zero-padded K beyond GDIM
__global__ __launch_bounds__(256) void k_pack_wg1(const float* __restrict__ Wg1,
                                                  unsigned short* __restrict__ out) {
    long i = (long)blockIdx.x * blockDim.x + threadIdx.x;
    if (i >= (long)4 * 384 * GPAD) return;
    int k = (int)(i % GPAD);
    int gn = (int)(i / GPAD);
    int n = gn % 384, g = gn / 384;
    float v = (k < GDIM) ? Wg1[((long)g * GDIM + k) * 384 + n] : 0.0f;
    out[i] = f2bf(v);
}

// Wsumt[n][k] (n<1024, ldk=256) = Wsum[k][n]
__global__ __launch_bounds__(256) void k_pack_wsum(const float* __restrict__ Wsum,
                                                   unsigned short* __restrict__ out) {
    long i = (long)blockIdx.x * blockDim.x + threadIdx.x;
    if (i >= (long)1024 * 256) return;
    int n = (int)(i >> 8), k = (int)(i & 255);
    out[i] = f2bf(Wsum[(long)k * 1024 + n]);
}

// ---------------------------------------------------------------------------
// GEMM 1: h1 = gelu(ctx_bf16[B x 1024] @ W1t + bias), bf16 out.
// Block = 8 waves x 16 rows (128 rows) x one 64-col group.
// B panel (64n x 32k) double-buffered in LDS via async loads.
// Grid: 128 rowblocks x 16 colgroups = 2048 blocks.
// ---------------------------------------------------------------------------
#define PANEL_STRIDE 40   // halves per n-row in LDS panel (80B, bank-spread)

__global__ __launch_bounds__(256) void k_gemm_h1(const unsigned short* __restrict__ A,
                                                 const unsigned short* __restrict__ Bt,
                                                 const float* __restrict__ bk1,
                                                 const float* __restrict__ bv1,
                                                 unsigned short* __restrict__ H) {
    __shared__ __align__(16) unsigned short sB[2][64 * PANEL_STRIDE];
    int tid = threadIdx.x;
    int lane = tid & 31, wv = tid >> 5;
    int rb = blockIdx.x & 127;
    int cg = blockIdx.x >> 7;                // 0..15
    int rowBase = (rb * 8 + wv) * 16;
    int colBase = cg * 64;
    int nlo = lane & 15;
    int khalf = (lane >> 4) << 4;
    int mbase = (lane >> 4) << 3;
    const unsigned short* pa = A + (long)(rowBase + nlo) * 1024 + ((lane >> 4) << 4);

    // staging: thread -> (row j = tid>>2, 16B chunk q = tid&3)
    int j = tid >> 2, q = tid & 3;
    const unsigned short* gsrc = Bt + (long)(colBase + j) * 1024 + q * 8;
    unsigned ldst0 = lds_addr(&sB[0][j * PANEL_STRIDE + q * 8]);
    unsigned ldst1 = lds_addr(&sB[1][j * PANEL_STRIDE + q * 8]);

    v8f acc[4];
#pragma unroll
    for (int n = 0; n < 4; ++n) acc[n] = vzero8();

    async_g2l_b128(ldst0, gsrc);             // stage k0 = 0 into buf 0
    int cur = 0;
    for (int k0 = 0; k0 < 1024; k0 += 32) {
        wait_asynccnt0();
        __syncthreads();
        if (k0 + 32 < 1024)
            async_g2l_b128(cur ? ldst0 : ldst1, gsrc + (k0 + 32));
        v16bf a = ld_a(pa + k0);
        const unsigned short* base = &sB[cur][0];
#pragma unroll
        for (int n = 0; n < 4; ++n) {
            v16bf b = ld_b(base + (n * 16 + nlo) * PANEL_STRIDE + khalf);
            acc[n] = WMMA_BF16(a, b, acc[n]);
        }
        cur ^= 1;
    }
#pragma unroll
    for (int n = 0; n < 4; ++n) {
        int col = colBase + n * 16 + nlo;
        float bias = (col < 512) ? bk1[col] : bv1[col - 512];
#pragma unroll
        for (int e = 0; e < 8; ++e) {
            float x = acc[n][e] + bias;
            H[(long)(rowBase + mbase + e) * 1024 + col] = f2bf(gelu_exact(x));
        }
    }
}

// ---------------------------------------------------------------------------
// GEMM 2: kv = h1 @ W2t + [bk2|bv2], f32 out [B x 384].
// Grid: 128 rowblocks x 6 colgroups = 768 blocks.
// ---------------------------------------------------------------------------
__global__ __launch_bounds__(256) void k_gemm_kv(const unsigned short* __restrict__ A,
                                                 const unsigned short* __restrict__ Bt,
                                                 const float* __restrict__ bk2,
                                                 const float* __restrict__ bv2,
                                                 float* __restrict__ KV) {
    __shared__ __align__(16) unsigned short sB[2][64 * PANEL_STRIDE];
    int tid = threadIdx.x;
    int lane = tid & 31, wv = tid >> 5;
    int rb = blockIdx.x % 128;
    int cg = blockIdx.x / 128;               // 0..5
    int rowBase = (rb * 8 + wv) * 16;
    int colBase = cg * 64;
    int nlo = lane & 15;
    int khalf = (lane >> 4) << 4;
    int mbase = (lane >> 4) << 3;
    const unsigned short* pa = A + (long)(rowBase + nlo) * 1024 + ((lane >> 4) << 4);

    int j = tid >> 2, q = tid & 3;
    const unsigned short* gsrc = Bt + (long)(colBase + j) * 1024 + q * 8;
    unsigned ldst0 = lds_addr(&sB[0][j * PANEL_STRIDE + q * 8]);
    unsigned ldst1 = lds_addr(&sB[1][j * PANEL_STRIDE + q * 8]);

    v8f acc[4];
#pragma unroll
    for (int n = 0; n < 4; ++n) acc[n] = vzero8();

    async_g2l_b128(ldst0, gsrc);
    int cur = 0;
    for (int k0 = 0; k0 < 1024; k0 += 32) {
        wait_asynccnt0();
        __syncthreads();
        if (k0 + 32 < 1024)
            async_g2l_b128(cur ? ldst0 : ldst1, gsrc + (k0 + 32));
        v16bf a = ld_a(pa + k0);
        const unsigned short* base = &sB[cur][0];
#pragma unroll
        for (int n = 0; n < 4; ++n) {
            v16bf b = ld_b(base + (n * 16 + nlo) * PANEL_STRIDE + khalf);
            acc[n] = WMMA_BF16(a, b, acc[n]);
        }
        cur ^= 1;
    }
#pragma unroll
    for (int n = 0; n < 4; ++n) {
        int col = colBase + n * 16 + nlo;
        float bias = (col < 128) ? bk2[col] : bv2[col - 128];
#pragma unroll
        for (int e = 0; e < 8; ++e)
            KV[(long)(rowBase + mbase + e) * 384 + col] = acc[n][e] + bias;
    }
}

// ---------------------------------------------------------------------------
// Row kernel: l2norm cand_key, slot match, spawn select, build bf16 feats row
// One wave per row, 8 rows per block, 2048 blocks.
// ---------------------------------------------------------------------------
__global__ __launch_bounds__(256) void k_rowmeta(const unsigned short* __restrict__ ctxb,
                                                 float* __restrict__ kv,
                                                 const float* __restrict__ keys,
                                                 const float* __restrict__ values,
                                                 const float* __restrict__ conf,
                                                 const float* __restrict__ age,
                                                 const float* __restrict__ alive,
                                                 const float* __restrict__ ledger,
                                                 unsigned short* __restrict__ feats,
                                                 int* __restrict__ best_idx,
                                                 int* __restrict__ spawn_idx,
                                                 float* __restrict__ has_live_f) {
    int lane = threadIdx.x & 31;
    int r = blockIdx.x * 8 + (threadIdx.x >> 5);
    float* kvr = kv + (long)r * 384;

    // --- normalize candidate key ---
    float ck[4]; float ss = 0.0f;
#pragma unroll
    for (int j = 0; j < 4; ++j) { ck[j] = kvr[lane * 4 + j]; ss += ck[j] * ck[j]; }
#pragma unroll
    for (int off = 1; off < 32; off <<= 1) ss += __shfl_xor(ss, off);
    float inv = 1.0f / fmaxf(sqrtf(ss), 1e-6f);
#pragma unroll
    for (int j = 0; j < 4; ++j) { ck[j] *= inv; kvr[lane * 4 + j] = ck[j]; }

    // --- slot cosine scores ---
    float sc[SS], al[SS];
#pragma unroll
    for (int s = 0; s < SS; ++s) {
        float a_s = alive[r * SS + s]; al[s] = a_s;
        float add = (1.0f - a_s) * 1e-4f;
        const float* kp = keys + (long)(r * SS + s) * 128;
        float sq = 0.0f, dt = 0.0f;
#pragma unroll
        for (int j = 0; j < 4; ++j) {
            float kkv = kp[lane * 4 + j] + add;
            sq += kkv * kkv; dt += kkv * ck[j];
        }
#pragma unroll
        for (int off = 1; off < 32; off <<= 1) { sq += __shfl_xor(sq, off); dt += __shfl_xor(dt, off); }
        float score = dt / fmaxf(sqrtf(sq), 1e-6f);
        sc[s] = (a_s > 0.0f) ? score : -1e4f;
    }
    float mx = sc[0]; int bi = 0;
#pragma unroll
    for (int s = 1; s < SS; ++s) if (sc[s] > mx) { mx = sc[s]; bi = s; }
    float hl = 0.0f; bool has_live = false;
#pragma unroll
    for (int s = 0; s < SS; ++s) { hl = fmaxf(hl, al[s]); has_live = has_live || (al[s] > 0.0f); }
    float best_score = has_live ? mx : 0.0f;
    if (!has_live) bi = 0;
    float ma = age[r * SS + bi];
    float cad = sigmoidf((ma - 8.0f) / 8.0f);
    float sur = 1.0f - best_score;

    int ic = -1;
#pragma unroll
    for (int s = 0; s < SS; ++s) if (al[s] < 0.5f && ic < 0) ic = s;
    float bestu = 1e30f; int ui = 0;
#pragma unroll
    for (int s = 0; s < SS; ++s) {
        float u = (s == bi) ? 1e4f : (conf[r * SS + s] - 0.01f * age[r * SS + s]);
        if (u < bestu) { bestu = u; ui = s; }
    }
    int si = (ic >= 0) ? ic : ui;

    if (lane == 0) { best_idx[r] = bi; spawn_idx[r] = si; has_live_f[r] = hl; }

    unsigned short* fr = feats + (long)r * GPAD;
#pragma unroll
    for (int j = 0; j < 32; ++j) fr[j * 32 + lane] = ctxb[(long)r * 1024 + j * 32 + lane];
#pragma unroll
    for (int j = 0; j < 4; ++j) fr[1024 + lane * 4 + j] = f2bf(ck[j]);
#pragma unroll
    for (int j = 0; j < 8; ++j) { int idx = j * 32 + lane; fr[1152 + idx] = f2bf(kvr[128 + idx]); }
    const float* mv = values + (long)(r * SS + bi) * 256;
#pragma unroll
    for (int j = 0; j < 8; ++j) { int idx = j * 32 + lane; fr[1408 + idx] = f2bf(mv[idx]); }
    if (lane == 0) {
        fr[1664] = f2bf(best_score);
        fr[1665] = f2bf(cad);
        fr[1666] = f2bf(sur);
        fr[1667] = f2bf(ledger[r * 2 + 0]);
        fr[1668] = f2bf(ledger[r * 2 + 1]);
    }
    if (lane < GPAD - GDIM) fr[GDIM + lane] = 0;
}

// ---------------------------------------------------------------------------
// Gate kernel: probs[b,g] = sigmoid( gelu(feats@Wg1[g]+bg1)·Wg2[g] + bg2[g] )
// Block = 8 waves x 16 rows (128 rows). 96 (gate,n-tile) iterations; the
// 16x1696 Wg1 strip for the next iteration is async-staged to LDS while the
// current strip feeds 53 WMMAs per wave. 128 blocks.
// ---------------------------------------------------------------------------
__global__ __launch_bounds__(256) void k_gates(const unsigned short* __restrict__ F,
                                               const unsigned short* __restrict__ Wg1t,
                                               const float* __restrict__ bg1,
                                               const float* __restrict__ Wg2,
                                               const float* __restrict__ bg2,
                                               float* __restrict__ probs) {
    __shared__ __align__(16) unsigned short sW[2][16 * GLDS];
    int tid = threadIdx.x;
    int lane = tid & 31, wv = tid >> 5;
    int rowBase = (blockIdx.x * 8 + wv) * 16;
    int nlo = lane & 15;
    int khalf = (lane >> 4) << 4;
    int mbase = (lane >> 4) << 3;
    const unsigned short* pa = F + (long)(rowBase + nlo) * GPAD + ((lane >> 4) << 4);

    // stage (g,nt) strip i: rows 16, GPAD halves each = 3392 16B chunks
    auto stageW = [&](int buf, int i) {
        const unsigned short* src = Wg1t + (long)(i * 16) * GPAD;  // i = g*24+nt
        for (int c = tid; c < 3392; c += 256) {
            int row = c / 212, off = (c % 212) * 8;
            async_g2l_b128(lds_addr(&sW[buf][row * GLDS + off]),
                           src + (long)row * GPAD + off);
        }
    };

    stageW(0, 0);
    float part[8];
#pragma unroll
    for (int e = 0; e < 8; ++e) part[e] = 0.0f;

    for (int i = 0; i < 96; ++i) {
        int g = i / 24, nt = i % 24;
        wait_asynccnt0();
        __syncthreads();
        if (i + 1 < 96) stageW((i + 1) & 1, i + 1);

        v8f acc = vzero8();
        const unsigned short* pb = &sW[i & 1][nlo * GLDS + khalf];
        for (int k0 = 0; k0 < GPAD; k0 += 32) {
            v16bf a = ld_a(pa + k0);
            acc = WMMA_BF16(a, ld_b(pb + k0), acc);
        }
        int col = nt * 16 + nlo;
        float bias = bg1[g * 384 + col];
        float wg = Wg2[g * 384 + col];
#pragma unroll
        for (int e = 0; e < 8; ++e)
            part[e] += wg * gelu_exact(acc[e] + bias);

        if (nt == 23) {  // finish this gate: reduce across 16-lane halves
#pragma unroll
            for (int off = 1; off < 16; off <<= 1)
#pragma unroll
                for (int e = 0; e < 8; ++e) part[e] += __shfl_xor(part[e], off);
            if (nlo == 0) {
                float bb = bg2[g];
#pragma unroll
                for (int e = 0; e < 8; ++e)
                    probs[(long)(rowBase + mbase + e) * 4 + g] = sigmoidf(part[e] + bb);
            }
#pragma unroll
            for (int e = 0; e < 8; ++e) part[e] = 0.0f;
        }
    }
}

// ---------------------------------------------------------------------------
// Update kernel: masks, scatter into slot memory, summary. One wave per row.
// ---------------------------------------------------------------------------
__global__ __launch_bounds__(256) void k_update(const float* __restrict__ kv,
                                                const float* __restrict__ keys,
                                                const float* __restrict__ values,
                                                const float* __restrict__ conf,
                                                const float* __restrict__ age,
                                                const float* __restrict__ alive,
                                                const float* __restrict__ probs,
                                                const int* __restrict__ best_idx,
                                                const int* __restrict__ spawn_idx,
                                                const float* __restrict__ has_live_f,
                                                float* __restrict__ out,
                                                unsigned short* __restrict__ sum_bf,
                                                float* __restrict__ promote_ws) {
    const long Bn = BB;
    int lane = threadIdx.x & 31;
    int r = blockIdx.x * 8 + (threadIdx.x >> 5);

    float rp = probs[(long)r * 4 + 0], sp = probs[(long)r * 4 + 1];
    float pp = probs[(long)r * 4 + 2], tp = probs[(long)r * 4 + 3];
    const float it = 4.0f;  // 1/TEMP
    float rr = sigmoidf((rp - 0.55f) * it) * has_live_f[r];
    float sr = sigmoidf((sp - 0.60f) * it);
    float den = 1.0f + rr + sr;
    float refresh = rr / den, spawn = sr / den;
    float promote = sigmoidf((pp - 0.50f) * it);
    float retire = sigmoidf((tp - 0.15f) * it);
    int bi = best_idx[r], si = spawn_idx[r];

    const float* kvr = kv + (long)r * 384;
    float ckv[4], cvv[8];
#pragma unroll
    for (int j = 0; j < 4; ++j) ckv[j] = kvr[lane * 4 + j];
#pragma unroll
    for (int j = 0; j < 8; ++j) cvv[j] = kvr[128 + j * 32 + lane];

    float sumv[8];
#pragma unroll
    for (int j = 0; j < 8; ++j) sumv[j] = 0.0f;
    float wsum = 0.0f;

    float* keys_o  = out + Bn * 1280;
    float* vals_o  = out + Bn * 2048;
    float* conf_o  = out + Bn * 3584;
    float* age_o   = out + Bn * 3590;
    float* alive_o = out + Bn * 3596;

#pragma unroll
    for (int s = 0; s < SS; ++s) {
        float a_s = alive[r * SS + s], c_s = conf[r * SS + s], g_s = age[r * SS + s];
        float rf  = (s == bi) ? refresh : 0.0f;
        float spn = (s == si) ? spawn   : 0.0f;
        float rt  = (s == bi) ? retire  : 0.0f;
        float wr = fminf(rf + spn, 1.0f);
        float om = 1.0f - wr;
#pragma unroll
        for (int j = 0; j < 4; ++j) {
            int idx = lane * 4 + j;
            keys_o[(long)(r * SS + s) * 128 + idx] =
                keys[(long)(r * SS + s) * 128 + idx] * om + wr * ckv[j];
        }
        float cn = (c_s * 0.995f * om + wr) * (1.0f - rt);
        float an = (g_s + a_s) * om;
        float ln = fminf(fmaxf(a_s * (1.0f - rt) + wr, 0.0f), 1.0f);
        float w_s = ln * cn;
        wsum += w_s;
#pragma unroll
        for (int j = 0; j < 8; ++j) {
            int idx = j * 32 + lane;
            float vn = values[(long)(r * SS + s) * 256 + idx] * om + wr * cvv[j];
            vals_o[(long)(r * SS + s) * 256 + idx] = vn;
            sumv[j] += vn * w_s;
        }
        if (lane == 0) {
            conf_o[r * SS + s] = cn;
            age_o[r * SS + s] = an;
            alive_o[r * SS + s] = ln;
        }
    }
    float invw = 1.0f / (wsum + 1e-6f);
    float* sum_o = out + Bn * 1024;
#pragma unroll
    for (int j = 0; j < 8; ++j) {
        int idx = j * 32 + lane;
        float sv = sumv[j] * invw;
        sum_o[(long)r * 256 + idx] = sv;
        sum_bf[(long)r * 256 + idx] = f2bf(sv);
    }
    if (lane == 0) promote_ws[r] = promote;
}

// ---------------------------------------------------------------------------
// Promoted kernel: z = summary@Wsum + bsum, layernorm, scale by promote.
// Block = 16 rows (8 waves, each 128 cols). 1024 blocks.
// ---------------------------------------------------------------------------
__global__ __launch_bounds__(256) void k_promoted(const unsigned short* __restrict__ Sb,
                                                  const unsigned short* __restrict__ Wt,
                                                  const float* __restrict__ bsum,
                                                  const float* __restrict__ g_ln,
                                                  const float* __restrict__ b_ln,
                                                  const float* __restrict__ promote,
                                                  float* __restrict__ out) {
    __shared__ float ps[16], ps2[16];
    int lane = threadIdx.x & 31;
    int wv = threadIdx.x >> 5;
    int rowBase = blockIdx.x * 16;
    int nlo = lane & 15;
    int khalf = (lane >> 4) << 4;
    int mbase = (lane >> 4) << 3;
    const unsigned short* pa = Sb + (long)(rowBase + nlo) * 256 + ((lane >> 4) << 4);

    v8f acc[8];
#pragma unroll
    for (int n = 0; n < 8; ++n) acc[n] = vzero8();

    for (int k0 = 0; k0 < 256; k0 += 32) {
        v16bf a = ld_a(pa + k0);
#pragma unroll
        for (int nt = 0; nt < 8; ++nt) {
            const unsigned short* pb = Wt + (long)(wv * 128 + nt * 16 + nlo) * 256 + k0 + khalf;
            acc[nt] = WMMA_BF16(a, ld_b(pb), acc[nt]);
        }
    }
#pragma unroll
    for (int nt = 0; nt < 8; ++nt) {
        int col = wv * 128 + nt * 16 + nlo;
        float bias = bsum[col];
#pragma unroll
        for (int e = 0; e < 8; ++e) acc[nt][e] += bias;
    }
    if (threadIdx.x < 16) { ps[threadIdx.x] = 0.0f; ps2[threadIdx.x] = 0.0f; }
    __syncthreads();
    float rs[8], rs2[8];
#pragma unroll
    for (int e = 0; e < 8; ++e) { rs[e] = 0.0f; rs2[e] = 0.0f; }
#pragma unroll
    for (int nt = 0; nt < 8; ++nt)
#pragma unroll
        for (int e = 0; e < 8; ++e) { float z = acc[nt][e]; rs[e] += z; rs2[e] += z * z; }
#pragma unroll
    for (int e = 0; e < 8; ++e) {
        atomicAdd(&ps[mbase + e], rs[e]);
        atomicAdd(&ps2[mbase + e], rs2[e]);
    }
    __syncthreads();
    if (threadIdx.x < 16) {
        float mu = ps[threadIdx.x] * (1.0f / 1024.0f);
        float var = ps2[threadIdx.x] * (1.0f / 1024.0f) - mu * mu;
        ps[threadIdx.x] = mu;
        ps2[threadIdx.x] = rsqrtf(var + 1e-5f);
    }
    __syncthreads();
#pragma unroll
    for (int nt = 0; nt < 8; ++nt) {
        int col = wv * 128 + nt * 16 + nlo;
        float gl = g_ln[col], bl = b_ln[col];
#pragma unroll
        for (int e = 0; e < 8; ++e) {
            int m = mbase + e;
            float z = acc[nt][e];
            out[(long)(rowBase + m) * 1024 + col] =
                promote[rowBase + m] * ((z - ps[m]) * ps2[m] * gl + bl);
        }
    }
}

// ---------------------------------------------------------------------------
// Host launcher
// ---------------------------------------------------------------------------
extern "C" void kernel_launch(void* const* d_in, const int* in_sizes, int n_in,
                              void* d_out, int out_size, void* d_ws, size_t ws_size,
                              hipStream_t stream) {
    const float* context = (const float*)d_in[0];
    const float* ledger  = (const float*)d_in[1];
    const float* keys    = (const float*)d_in[2];
    const float* values  = (const float*)d_in[3];
    const float* conf    = (const float*)d_in[4];
    const float* age     = (const float*)d_in[5];
    const float* alive   = (const float*)d_in[6];
    const float* Wk1     = (const float*)d_in[7];
    const float* bk1     = (const float*)d_in[8];
    const float* Wk2     = (const float*)d_in[9];
    const float* bk2     = (const float*)d_in[10];
    const float* Wv1     = (const float*)d_in[11];
    const float* bv1     = (const float*)d_in[12];
    const float* Wv2     = (const float*)d_in[13];
    const float* bv2     = (const float*)d_in[14];
    const float* Wg1     = (const float*)d_in[15];
    const float* bg1     = (const float*)d_in[16];
    const float* Wg2     = (const float*)d_in[17];
    const float* bg2     = (const float*)d_in[18];
    const float* Wsum    = (const float*)d_in[19];
    const float* bsum    = (const float*)d_in[20];
    const float* g_ln    = (const float*)d_in[21];
    const float* b_ln    = (const float*)d_in[22];
    float* out = (float*)d_out;

    char* base = (char*)d_ws;
    size_t off = 0;
    auto alloc = [&](size_t bytes) -> char* {
        char* p = base + off;
        off += (bytes + 255) & ~(size_t)255;
        return p;
    };
    unsigned short* ws_ctxb  = (unsigned short*)alloc((size_t)BB * 1024 * 2);
    unsigned short* ws_h1    = (unsigned short*)alloc((size_t)BB * 1024 * 2);
    unsigned short* ws_w1t   = (unsigned short*)alloc((size_t)1024 * 1024 * 2);
    unsigned short* ws_w2t   = (unsigned short*)alloc((size_t)384 * 1024 * 2);
    unsigned short* ws_wg1t  = (unsigned short*)alloc((size_t)4 * 384 * GPAD * 2);
    unsigned short* ws_wsumt = (unsigned short*)alloc((size_t)1024 * 256 * 2);
    float*          ws_kv    = (float*)alloc((size_t)BB * 384 * 4);
    unsigned short* ws_feats = (unsigned short*)alloc((size_t)BB * GPAD * 2);
    float*          ws_probs = (float*)alloc((size_t)BB * 4 * 4);
    int*            ws_bidx  = (int*)alloc((size_t)BB * 4);
    int*            ws_sidx  = (int*)alloc((size_t)BB * 4);
    float*          ws_hl    = (float*)alloc((size_t)BB * 4);
    float*          ws_prom  = (float*)alloc((size_t)BB * 4);
    unsigned short* ws_sumbf = (unsigned short*)alloc((size_t)BB * 256 * 2);
    (void)ws_size; (void)in_sizes; (void)n_in; (void)out_size;

    {
        long n = (long)BB * 1024;
        k_cvt_bf16<<<(unsigned)((n + 255) / 256), 256, 0, stream>>>(context, ws_ctxb, n);
    }
    k_pack_w1<<<(1024 * 1024 + 255) / 256, 256, 0, stream>>>(Wk1, Wv1, ws_w1t);
    k_pack_w2<<<(384 * 1024 + 255) / 256, 256, 0, stream>>>(Wk2, Wv2, ws_w2t);
    k_pack_wg1<<<(4 * 384 * GPAD + 255) / 256, 256, 0, stream>>>(Wg1, ws_wg1t);
    k_pack_wsum<<<(1024 * 256 + 255) / 256, 256, 0, stream>>>(Wsum, ws_wsumt);

    k_gemm_h1<<<2048, 256, 0, stream>>>(ws_ctxb, ws_w1t, bk1, bv1, ws_h1);
    k_gemm_kv<<<768, 256, 0, stream>>>(ws_h1, ws_w2t, bk2, bv2, ws_kv);
    k_rowmeta<<<2048, 256, 0, stream>>>(ws_ctxb, ws_kv, keys, values, conf, age, alive,
                                        ledger, ws_feats, ws_bidx, ws_sidx, ws_hl);
    k_gates<<<128, 256, 0, stream>>>(ws_feats, ws_wg1t, bg1, Wg2, bg2, ws_probs);
    k_update<<<2048, 256, 0, stream>>>(ws_kv, keys, values, conf, age, alive, ws_probs,
                                       ws_bidx, ws_sidx, ws_hl, out, ws_sumbf, ws_prom);
    k_promoted<<<1024, 256, 0, stream>>>(ws_sumbf, ws_wsumt, bsum, g_ln, b_ln, ws_prom, out);
}
